// ImageTriplaneGenerator_15144054686224
// MI455X (gfx1250) — compile-verified, matrix operand order checked by reading, not run on hardware
//
#include <hip/hip_runtime.h>

typedef __attribute__((ext_vector_type(16))) _Float16 v16h;
typedef __attribute__((ext_vector_type(8)))  _Float16 v8h;
typedef __attribute__((ext_vector_type(8)))  float    v8f;

namespace {
constexpr int Cc    = 196;              // channels
constexpr int CP    = 208;              // padded out-channels (13*16)
constexpr int AS    = 232;              // row stride (halfs) for A/W/cbar (>=224, bank-friendly)
constexpr int KS    = 200;              // K-result LDS row stride (halfs)
constexpr int POSB  = 16;               // query positions per block
constexpr int ROWS  = POSB * 8;         // 128 context rows
constexpr int HEADS = 4;
constexpr int NV    = 8;
constexpr int DH    = 49;               // head dim
constexpr int NT    = 13;               // N tiles (208/16)
constexpr int KSTP  = 7;                // K steps of 32 (224)
constexpr int LSEQ  = 16384;

// LDS layout (bytes), all offsets 16B aligned
constexpr int OFF_W  = 0;                          // Wk/Wv buffer: 208*232*2 = 96512
constexpr int OFF_A  = OFF_W + CP * AS * 2;        // context f16: 128*232*2 = 59392
constexpr int OFF_K  = OFF_A + ROWS * AS * 2;      // k results f16: 128*200*2 = 51200
constexpr int OFF_Q  = OFF_K + ROWS * KS * 2;      // normalized q f32: 16*196*4 = 12544
constexpr int OFF_S  = OFF_Q + POSB * Cc * 4;      // raw scores: 16*4*8*4 = 2048
constexpr int OFF_P  = OFF_S + POSB * HEADS * NV * 4; // probs: 2048
constexpr int OFF_CB = OFF_P + POSB * HEADS * NV * 4; // cbar f16: 64*232*2 = 29696
constexpr int SMEM_BYTES = OFF_CB + 64 * AS * 2;   // total 253440
} // namespace

__global__ __launch_bounds__(256, 1)
void triplane_attn(const float* __restrict__ query, const float* __restrict__ ctx,
                   const float* __restrict__ Wk, const float* __restrict__ bk,
                   const float* __restrict__ Wv, const float* __restrict__ bv,
                   const float* __restrict__ gamma, const float* __restrict__ beta,
                   float* __restrict__ out, float* __restrict__ attn)
{
    extern __shared__ char smem[];
    _Float16* Wl = (_Float16*)(smem + OFF_W);
    _Float16* Al = (_Float16*)(smem + OFF_A);
    _Float16* Kl = (_Float16*)(smem + OFF_K);
    float*    Ql = (float*)   (smem + OFF_Q);
    float*    Sl = (float*)   (smem + OFF_S);
    float*    Pl = (float*)   (smem + OFF_P);
    _Float16* Cb = (_Float16*)(smem + OFF_CB);

    const int  tid   = threadIdx.x;
    const int  wave  = tid >> 5;
    const int  lane  = tid & 31;
    const int  ln16  = lane & 15;
    const int  lhalf = lane >> 4;
    const long p0    = (long)blockIdx.x * POSB;

    // ---------------- zero LDS pad regions ----------------
    for (int i = tid; i < ROWS * (AS - Cc); i += 256)
        Al[(i / (AS - Cc)) * AS + Cc + (i % (AS - Cc))] = (_Float16)0.f;
    for (int i = tid; i < Cc * (AS - Cc); i += 256)
        Wl[(i / (AS - Cc)) * AS + Cc + (i % (AS - Cc))] = (_Float16)0.f;
    for (int i = tid; i < (CP - Cc) * AS; i += 256)
        Wl[Cc * AS + i] = (_Float16)0.f;
    for (int i = tid; i < 64 * (AS - Cc); i += 256)
        Cb[(i / (AS - Cc)) * AS + Cc + (i % (AS - Cc))] = (_Float16)0.f;

    // ---------------- load Wk (f32 -> f16) ----------------
    for (int i = tid; i < Cc * (Cc / 4); i += 256) {
        int r = i / 49, c = (i % 49) * 4;
        float4 w = ((const float4*)Wk)[i];
        _Float16* d = &Wl[r * AS + c];
        d[0] = (_Float16)w.x; d[1] = (_Float16)w.y; d[2] = (_Float16)w.z; d[3] = (_Float16)w.w;
    }
    // ---------------- load context block (f32 -> f16) ----------------
    const float* cbase = ctx + p0 * NV * Cc;
    for (int i = tid; i < ROWS * 49; i += 256) {
        int r = i / 49, c = (i % 49) * 4;
        float4 w = *(const float4*)(cbase + (long)r * Cc + c);
        _Float16* d = &Al[r * AS + c];
        d[0] = (_Float16)w.x; d[1] = (_Float16)w.y; d[2] = (_Float16)w.z; d[3] = (_Float16)w.w;
    }
    // ---------------- load raw query ----------------
    for (int i = tid; i < POSB * 49; i += 256) {
        int r = i / 49, c = (i % 49) * 4;
        float4 w = *(const float4*)(query + (p0 + r) * Cc + c);
        float* d = &Ql[r * Cc + c];
        d[0] = w.x; d[1] = w.y; d[2] = w.z; d[3] = w.w;
    }
    __syncthreads();

    // ---------------- layernorm(q): 16 lanes per position ----------------
    {
        int pos = tid >> 4, j = tid & 15;
        float s = 0.f, s2 = 0.f;
        for (int c = j; c < Cc; c += 16) { float x = Ql[pos * Cc + c]; s += x; s2 += x * x; }
        for (int off = 8; off >= 1; off >>= 1) {
            s  += __shfl_xor(s,  off, 32);
            s2 += __shfl_xor(s2, off, 32);
        }
        float mu  = s * (1.f / Cc);
        float var = s2 * (1.f / Cc) - mu * mu;
        float inv = rsqrtf(var + 1e-5f);
        for (int c = j; c < Cc; c += 16) {
            float x = Ql[pos * Cc + c];
            Ql[pos * Cc + c] = (x - mu) * inv * gamma[c] + beta[c];
        }
    }

    // ---------------- K projection GEMM: wave = M-tile ----------------
    {
        const int mt   = wave;
        const int arow = mt * 16 + ln16;
        v16h afr[KSTP];
#pragma unroll
        for (int ks = 0; ks < KSTP; ks++) {
            v8h lo = *(const v8h*)&Al[arow * AS + ks * 32 + lhalf * 8];
            v8h up = *(const v8h*)&Al[arow * AS + ks * 32 + 16 + lhalf * 8];
            v16h a;
#pragma unroll
            for (int q = 0; q < 8; q++) { a[q] = lo[q]; a[q + 8] = up[q]; }
            afr[ks] = a;
        }
        for (int nt = 0; nt < NT; nt++) {
            const int wrow = nt * 16 + ln16;
            v8f acc = {};
#pragma unroll
            for (int ks = 0; ks < KSTP; ks++) {
                v8h lo = *(const v8h*)&Wl[wrow * AS + ks * 32 + lhalf * 8];
                v8h up = *(const v8h*)&Wl[wrow * AS + ks * 32 + 16 + lhalf * 8];
                v16h b;
#pragma unroll
                for (int q = 0; q < 8; q++) { b[q] = lo[q]; b[q + 8] = up[q]; }
                acc = __builtin_amdgcn_wmma_f32_16x16x32_f16(
                          false, afr[ks], false, b, (short)0, acc, false, false);
            }
            const int c = nt * 16 + ln16;
            if (c < Cc) {
                const float bias = bk[c];
#pragma unroll
                for (int q = 0; q < 8; q++) {
                    int row = mt * 16 + q + 8 * lhalf;   // D: M = vgpr + 8*(lane>>4)
                    Kl[row * KS + c] = (_Float16)(acc[q] + bias);
                }
            }
        }
    }
    __syncthreads();

    // ---------------- attention scores ----------------
    for (int it = tid; it < POSB * HEADS * NV; it += 256) {
        int pos = it >> 5, hh = (it >> 3) & 3, n = it & 7;
        const float*    qp = &Ql[pos * Cc + hh * DH];
        const _Float16* kp = &Kl[(pos * NV + n) * KS + hh * DH];
        float s = 0.f;
        for (int j = 0; j < DH; j++) s += qp[j] * (float)kp[j];
        Sl[it] = s * (1.f / 7.f);   // 1/sqrt(49)
    }
    // ---------------- stream Wv into the (now free) W buffer ----------------
    for (int i = tid; i < Cc * (Cc / 4); i += 256) {
        int r = i / 49, c = (i % 49) * 4;
        float4 w = ((const float4*)Wv)[i];
        _Float16* d = &Wl[r * AS + c];
        d[0] = (_Float16)w.x; d[1] = (_Float16)w.y; d[2] = (_Float16)w.z; d[3] = (_Float16)w.w;
    }
    __syncthreads();

    // ---------------- softmax + attn_prob output ----------------
    if (tid < POSB * HEADS) {
        int pos = tid >> 2, hh = tid & 3;
        const float* s = &Sl[(pos * HEADS + hh) * NV];
        float mx = s[0];
        for (int n = 1; n < NV; n++) mx = fmaxf(mx, s[n]);
        float e[NV]; float sum = 0.f;
        for (int n = 0; n < NV; n++) { e[n] = __expf(s[n] - mx); sum += e[n]; }
        float inv = 1.f / sum;
        long p = p0 + pos, b = p >> 14, l = p & (LSEQ - 1);
        float* ap = attn + ((b * HEADS + hh) * (long)LSEQ + l) * NV;
        float* pp = &Pl[(pos * HEADS + hh) * NV];
        for (int n = 0; n < NV; n++) { float pr = e[n] * inv; pp[n] = pr; ap[n] = pr; }
    }
    __syncthreads();

    // ---------------- weighted context cbar[r = h*16+pos][c] ----------------
    {
        int r = tid >> 2, g = tid & 3;
        int hh = r >> 4, pos = r & 15;
        const float* pp = &Pl[(pos * HEADS + hh) * NV];
        float pr[NV];
        for (int n = 0; n < NV; n++) pr[n] = pp[n];
        const _Float16* cx = &Al[(pos * NV) * AS];
        for (int c = g * DH; c < g * DH + DH; c++) {
            float acc = 0.f;
            for (int n = 0; n < NV; n++) acc += pr[n] * (float)cx[n * AS + c];
            Cb[r * AS + c] = (_Float16)acc;
        }
    }
    __syncthreads();

    // ---------------- V GEMM: M-tile = head (wave>>1), 2 N-tiles per wave ----------------
    {
        const int hh   = wave >> 1;
        const int arow = hh * 16 + ln16;
        v16h afr[KSTP];
#pragma unroll
        for (int ks = 0; ks < KSTP; ks++) {
            v8h lo = *(const v8h*)&Cb[arow * AS + ks * 32 + lhalf * 8];
            v8h up = *(const v8h*)&Cb[arow * AS + ks * 32 + 16 + lhalf * 8];
            v16h a;
#pragma unroll
            for (int q = 0; q < 8; q++) { a[q] = lo[q]; a[q + 8] = up[q]; }
            afr[ks] = a;
        }
#pragma unroll
        for (int tt = 0; tt < 2; tt++) {
            const int t    = wave * 2 + tt;
            const int nt   = (DH * hh) / 16 + (t & 3);  // 4 N-tiles covering head hh's channels
            const int wrow = nt * 16 + ln16;
            v8f acc = {};
#pragma unroll
            for (int ks = 0; ks < KSTP; ks++) {
                v8h lo = *(const v8h*)&Wl[wrow * AS + ks * 32 + lhalf * 8];
                v8h up = *(const v8h*)&Wl[wrow * AS + ks * 32 + 16 + lhalf * 8];
                v16h b;
#pragma unroll
                for (int q = 0; q < 8; q++) { b[q] = lo[q]; b[q + 8] = up[q]; }
                acc = __builtin_amdgcn_wmma_f32_16x16x32_f16(
                          false, afr[ks], false, b, (short)0, acc, false, false);
            }
            const int c = nt * 16 + ln16;
            if (c < Cc && c >= DH * hh && c < DH * hh + DH) {
                const float bias = bv[c];
#pragma unroll
                for (int q = 0; q < 8; q++) {
                    int pos = q + 8 * lhalf;
                    out[(p0 + pos) * Cc + c] = acc[q] + bias;
                }
            }
        }
    }
}

extern "C" void kernel_launch(void* const* d_in, const int* in_sizes, int n_in,
                              void* d_out, int out_size, void* d_ws, size_t ws_size,
                              hipStream_t stream) {
    const float* query = (const float*)d_in[0];
    const float* ctx   = (const float*)d_in[1];
    const float* Wk    = (const float*)d_in[2];
    const float* bk    = (const float*)d_in[3];
    const float* Wv    = (const float*)d_in[4];
    const float* bv    = (const float*)d_in[5];
    const float* gamma = (const float*)d_in[6];
    const float* beta  = (const float*)d_in[7];

    float* out  = (float*)d_out;
    const size_t positions = (size_t)in_sizes[0] / Cc;   // B*L = 32768
    float* attn = out + positions * Cc;                  // second tuple output

    (void)d_ws; (void)ws_size; (void)n_in; (void)out_size;

    hipFuncSetAttribute((const void*)triplane_attn,
                        hipFuncAttributeMaxDynamicSharedMemorySize, SMEM_BYTES);

    const int blocks = (int)(positions / POSB);          // 2048
    triplane_attn<<<blocks, 256, SMEM_BYTES, stream>>>(
        query, ctx, Wk, bk, Wv, bv, gamma, beta, out, attn);
}
